// GRU_2267742732465
// MI455X (gfx1250) — matrix operand order repaired and checked by reading, not compiled, not run on hardware
//
#include <hip/hip_runtime.h>

#define SEQ    512
#define BATCH  256
#define EDIM   512
#define LDIM   1024
#define KTOT   1536   // E + L
#define NZR    2048   // [z | r] combined gate columns
#define KTILES 48     // KTOT / 32

typedef __bf16 bf16;
typedef __attribute__((ext_vector_type(16))) __bf16 v16bf;
typedef __attribute__((ext_vector_type(8)))  float  v8f;

// ---- fp32 -> bf16, round-to-nearest-even ----
__device__ __forceinline__ bf16 f2bf(float f) {
    unsigned int u = __builtin_bit_cast(unsigned int, f);
    u = (u + 0x7FFFu + ((u >> 16) & 1u)) >> 16;
    unsigned short s = (unsigned short)u;
    return __builtin_bit_cast(bf16, s);
}

// ---- WMMA fragment-slot helpers (CDNA5 ISA 7.12.2 layouts) ----
// A matrix (16-bit, 16x32 MxK): lane = m + 16*half, half selects K groups of 8.
__device__ __forceinline__ int a_slot(int b, int k) {
    int bt = b >> 4, kt = k >> 5, kk = k & 31;
    int half = (kk & 15) >> 3;
    int idx  = ((kk >> 4) << 3) | (kk & 7);
    int lane = (b & 15) | (half << 4);
    return ((bt * KTILES + kt) << 9) + (lane << 4) + idx;
}
// B matrix (32x16 KxN): lane = n + 16*half, half selects K groups of 8 (same as A).
__device__ __forceinline__ int b_slot(int k, int n) {
    int nt = n >> 4, kt = k >> 5, kk = k & 31;
    int half = (kk & 15) >> 3;
    int idx  = ((kk >> 4) << 3) | (kk & 7);
    int lane = (n & 15) | (half << 4);
    return ((nt * KTILES + kt) << 9) + (lane << 4) + idx;
}

// ---- embedding gather body: xe(t) -> bf16 swizzled into both A buffers ----
__device__ __forceinline__ void gather_body(const int* __restrict__ x,
                                            const float* __restrict__ emb, int t,
                                            bf16* __restrict__ A1x,
                                            bf16* __restrict__ A2x,
                                            int tid, int nthreads) {
    // BATCH*EDIM = 131072 elements
    for (int e = tid; e < BATCH * EDIM; e += nthreads) {
        int b = e >> 9, k = e & 511;
        int tok = x[(size_t)t * BATCH + b];
        bf16 v = f2bf(emb[(size_t)tok * EDIM + k]);
        int s = a_slot(b, k);
        A1x[s] = v;
        A2x[s] = v;
    }
}

// ---- one-time: convert weights to swizzled bf16 ----
__global__ void __launch_bounds__(256)
k_convert_weights(const float* __restrict__ Wz, const float* __restrict__ Wr,
                  const float* __restrict__ Wh, bf16* __restrict__ Bzr,
                  bf16* __restrict__ Bh) {
    int tid = blockIdx.x * 256 + threadIdx.x;
    if (tid >= KTOT * 3072) return;
    int k = tid / 3072;
    int n = tid % 3072;
    if (n < NZR) {
        float v = (n < LDIM) ? Wz[k * LDIM + n] : Wr[k * LDIM + (n - LDIM)];
        Bzr[b_slot(k, n)] = f2bf(v);
    } else {
        int nh = n - NZR;
        Bh[b_slot(k, nh)] = f2bf(Wh[k * LDIM + nh]);
    }
}

// ---- one-time: h0 -> hbuf (f32) and A1[0] h-region (bf16 swizzled) ----
__global__ void __launch_bounds__(256)
k_init_h(const float* __restrict__ h0, float* __restrict__ hbuf,
         bf16* __restrict__ A1) {
    int tid = blockIdx.x * 256 + threadIdx.x;   // BATCH*LDIM
    int b = tid >> 10, n = tid & 1023;
    float v = h0[tid];
    hbuf[tid] = v;
    A1[a_slot(b, EDIM + n)] = f2bf(v);
}

// ---- standalone gather for t = 0 ----
__global__ void __launch_bounds__(256)
k_gather0(const int* __restrict__ x, const float* __restrict__ emb,
          bf16* __restrict__ A1, bf16* __restrict__ A2) {
    gather_body(x, emb, 0, A1, A2,
                blockIdx.x * 256 + threadIdx.x, gridDim.x * 256);
}

// ---- WMMA core: one wave computes a 32x32 C block (2x2 tiles) over K=1536 ----
__device__ __forceinline__ void wmma_core(const bf16* __restrict__ A,
                                          const bf16* __restrict__ Bw,
                                          int bt0, int nt0, v8f acc[2][2]) {
    const int lane = threadIdx.x & 31;
    const bf16* a0p = A  + (( bt0      * KTILES) << 9) + (lane << 4);
    const bf16* a1p = A  + (((bt0 + 1) * KTILES) << 9) + (lane << 4);
    const bf16* b0p = Bw + (( nt0      * KTILES) << 9) + (lane << 4);
    const bf16* b1p = Bw + (((nt0 + 1) * KTILES) << 9) + (lane << 4);
#pragma unroll 4
    for (int kt = 0; kt < KTILES; ++kt) {
        v16bf a0 = *(const v16bf*)(a0p + (kt << 9));
        v16bf a1 = *(const v16bf*)(a1p + (kt << 9));
        v16bf b0 = *(const v16bf*)(b0p + (kt << 9));
        v16bf b1 = *(const v16bf*)(b1p + (kt << 9));
        acc[0][0] = __builtin_amdgcn_wmma_f32_16x16x32_bf16(false, a0, false, b0, (short)0, acc[0][0], false, false);
        acc[0][1] = __builtin_amdgcn_wmma_f32_16x16x32_bf16(false, a0, false, b1, (short)0, acc[0][1], false, false);
        acc[1][0] = __builtin_amdgcn_wmma_f32_16x16x32_bf16(false, a1, false, b0, (short)0, acc[1][0], false, false);
        acc[1][1] = __builtin_amdgcn_wmma_f32_16x16x32_bf16(false, a1, false, b1, (short)0, acc[1][1], false, false);
    }
}

// ---- GEMM 1 + fused gather(t+1):
//   z==0 blocks: [xe|h](p) @ [Wz|Wr]; epilogue: z->zbuf, r -> r*h bf16 into A2(p)
//   z==1 blocks: gather xe(t+1) into A1(p^1), A2(p^1)  (concurrent, disjoint buffers)
__global__ void __launch_bounds__(128)
k_gemm_zr(const bf16* __restrict__ A1c, const bf16* __restrict__ Bzr,
          const float* __restrict__ bz, const float* __restrict__ br,
          const float* __restrict__ hbuf, float* __restrict__ zbuf,
          bf16* __restrict__ A2c,
          bf16* __restrict__ A1n, bf16* __restrict__ A2n,
          const int* __restrict__ x, const float* __restrict__ emb, int tn) {
    if (blockIdx.z == 1) {
        if (tn < 0) return;
        int bid = blockIdx.y * gridDim.x + blockIdx.x;          // 0..127
        gather_body(x, emb, tn, A1n, A2n,
                    bid * 128 + threadIdx.x, gridDim.x * gridDim.y * 128);
        return;
    }
    int w = threadIdx.x >> 5, lane = threadIdx.x & 31;
    int bt0 = blockIdx.y * 4 + (w >> 1) * 2;
    int nt0 = blockIdx.x * 4 + (w & 1) * 2;
    v8f acc[2][2] = {};
    wmma_core(A1c, Bzr, bt0, nt0, acc);
    int nlo = lane & 15, half = lane >> 4;
#pragma unroll
    for (int i = 0; i < 2; ++i)
#pragma unroll
        for (int j = 0; j < 2; ++j) {
            int n = ((nt0 + j) << 4) + nlo;
#pragma unroll
            for (int v = 0; v < 8; ++v) {
                int b = ((bt0 + i) << 4) + v + (half << 3);
                float a = acc[i][j][v];
                if (n < LDIM) {
                    float z = 1.f / (1.f + __expf(-(a + bz[n])));
                    zbuf[b * LDIM + n] = z;
                } else {
                    int nr = n - LDIM;
                    float r = 1.f / (1.f + __expf(-(a + br[nr])));
                    A2c[a_slot(b, EDIM + nr)] = f2bf(r * hbuf[b * LDIM + nr]);
                }
            }
        }
}

// ---- GEMM 2: [xe|r*h](p) @ Wh; epilogue: h' = (1-z)h + z*tanh -> hbuf, out, A1(p^1) ----
__global__ void __launch_bounds__(128)
k_gemm_h(const bf16* __restrict__ A2c, const bf16* __restrict__ Bh,
         const float* __restrict__ bh, const float* __restrict__ zbuf,
         float* __restrict__ hbuf, bf16* __restrict__ A1n,
         float* __restrict__ out_t) {
    int w = threadIdx.x >> 5, lane = threadIdx.x & 31;
    int bt0 = blockIdx.y * 4 + (w >> 1) * 2;
    int nt0 = blockIdx.x * 4 + (w & 1) * 2;
    v8f acc[2][2] = {};
    wmma_core(A2c, Bh, bt0, nt0, acc);
    int nlo = lane & 15, half = lane >> 4;
#pragma unroll
    for (int i = 0; i < 2; ++i)
#pragma unroll
        for (int j = 0; j < 2; ++j) {
            int n = ((nt0 + j) << 4) + nlo;
#pragma unroll
            for (int v = 0; v < 8; ++v) {
                int b = ((bt0 + i) << 4) + v + (half << 3);
                float cand = tanhf(acc[i][j][v] + bh[n]);
                float z = zbuf[b * LDIM + n];
                float hn = (1.f - z) * hbuf[b * LDIM + n] + z * cand;
                hbuf[b * LDIM + n]  = hn;           // h for next step (f32)
                out_t[b * LDIM + n] = hn;           // hs[t]
                A1n[a_slot(b, EDIM + n)] = f2bf(hn); // next step's A1 operand
            }
        }
}

extern "C" void kernel_launch(void* const* d_in, const int* in_sizes, int n_in,
                              void* d_out, int out_size, void* d_ws, size_t ws_size,
                              hipStream_t stream) {
    const int*   x   = (const int*)  d_in[0];
    const float* h0  = (const float*)d_in[1];
    const float* emb = (const float*)d_in[2];
    const float* Wz  = (const float*)d_in[3];
    const float* bz  = (const float*)d_in[4];
    const float* Wr  = (const float*)d_in[5];
    const float* br  = (const float*)d_in[6];
    const float* Wh  = (const float*)d_in[7];
    const float* bh  = (const float*)d_in[8];
    float* out = (float*)d_out;

    char* ws = (char*)d_ws;
    size_t off = 0;
    auto carve = [&](size_t bytes) { void* p = ws + off; off += (bytes + 255) & ~(size_t)255; return p; };
    bf16*  Bzr   = (bf16*) carve((size_t)KTOT * NZR  * 2);   // 6 MB
    bf16*  Bh    = (bf16*) carve((size_t)KTOT * LDIM * 2);   // 3 MB
    bf16*  A1b[2], *A2b[2];
    A1b[0] = (bf16*)carve((size_t)BATCH * KTOT * 2);         // [xe | h], parity 0
    A1b[1] = (bf16*)carve((size_t)BATCH * KTOT * 2);         // parity 1
    A2b[0] = (bf16*)carve((size_t)BATCH * KTOT * 2);         // [xe | r*h], parity 0
    A2b[1] = (bf16*)carve((size_t)BATCH * KTOT * 2);         // parity 1
    float* zbuf  = (float*)carve((size_t)BATCH * LDIM * 4);
    float* hbuf  = (float*)carve((size_t)BATCH * LDIM * 4);

    // one-time prep (re-run every call for determinism)
    k_convert_weights<<<(KTOT * 3072 + 255) / 256, 256, 0, stream>>>(Wz, Wr, Wh, Bzr, Bh);
    k_init_h<<<(BATCH * LDIM) / 256, 256, 0, stream>>>(h0, hbuf, A1b[0]);
    k_gather0<<<128, 256, 0, stream>>>(x, emb, A1b[0], A2b[0]);

    dim3 g1(NZR / 64, BATCH / 64, 2);   // z=0: 32x4 GEMM blocks; z=1: gather(t+1)
    dim3 g2(LDIM / 64, BATCH / 64);     // 16x4 GEMM blocks
    for (int t = 0; t < SEQ; ++t) {
        int p = t & 1;
        int tn = (t + 1 < SEQ) ? (t + 1) : -1;
        k_gemm_zr<<<g1, 128, 0, stream>>>(A1b[p], Bzr, bz, br, hbuf, zbuf,
                                          A2b[p], A1b[p ^ 1], A2b[p ^ 1],
                                          x, emb, tn);
        k_gemm_h <<<g2, 128, 0, stream>>>(A2b[p], Bh, bh, zbuf, hbuf, A1b[p ^ 1],
                                          out + (size_t)t * BATCH * LDIM);
    }
}
